// OptimizedMOEImproved_11390253269276
// MI455X (gfx1250) — compile-verified, matrix operand order checked by reading, not run on hardware
//
#include <hip/hip_runtime.h>
#include <hip/hip_bf16.h>

// ---------------- CDNA5 WMMA types ----------------
typedef __attribute__((ext_vector_type(16))) __bf16 v16bf;
typedef __attribute__((ext_vector_type(8)))  __bf16 v8bf;
typedef __attribute__((ext_vector_type(8)))  float  v8f;

static __device__ __forceinline__ float silu_f(float v) { return v / (1.0f + __expf(-v)); }

static __device__ __forceinline__ unsigned short f2bf(float f) {
    __bf16 h = (__bf16)f;                       // RTNE f32->bf16 convert
    return __builtin_bit_cast(unsigned short, h);
}
static __device__ __forceinline__ float bf2f(unsigned short u) {
    return __builtin_bit_cast(float, ((unsigned)u) << 16);  // exact
}

// ---------------- problem constants ----------------
#define BN_     8
#define CIN     256
#define HW      64
#define M_CH    128          // init_ch == new_ch
#define KDIM    2304         // 9 taps * 256 ci   (k = tap*256 + ci)
#define KCHUNK  64           // two WMMA K-steps per stage; 64 | 256 so tap is chunk-uniform
#define COUT    256
#define NEXP    5            // 4 routed + 1 shared
#define OUTROWS 4            // output rows per workgroup band
#define Y1R     6            // y1 rows incl. depthwise halo
#define NPOS    (Y1R*64)     // 384 positions per band
#define NPASS   3            // two y1 rows (128 positions) per pass

// ================= small kernels =================
__global__ void zero_out_k(float* out, int n) {
    int i = blockIdx.x * 256 + threadIdx.x;
    if (i < n) out[i] = 0.0f;
}

// x [8,256,64,64] f32 (NCHW) -> xT [8,64,64,256] bf16 (channels-last).
__global__ void xpose_k(const float* __restrict__ x, unsigned short* __restrict__ xT) {
    int i = blockIdx.x * 256 + threadIdx.x;          // [b][p][ci]
    if (i >= BN_ * HW * HW * CIN) return;
    int ci = i & 255;
    int p  = (i >> 8) & 4095;
    int b  = i >> 20;
    xT[i] = f2bf(x[((size_t)b * CIN + ci) * (HW * HW) + p]);
}

// Pack primary conv weights (f32 OIHW) -> bf16 A matrices [5][128][KDIM]
// with PERMUTED k-order: k = tap*256 + ci  (tap = kh*3+kw).
__global__ void pack_weights_k(const float* __restrict__ We_p,
                               const float* __restrict__ Ws_p,
                               unsigned short* __restrict__ Apack) {
    const int per = M_CH * KDIM;
    int o = blockIdx.x * 256 + threadIdx.x;
    if (o >= NEXP * per) return;
    int mg  = o / KDIM;                    // 0..639  (= e*128 + m)
    int r   = o - mg * KDIM;
    int tap = r >> 8, ci = r & 255;
    float v = (mg < 4 * M_CH) ? We_p[mg * KDIM + ci * 9 + tap]
                              : Ws_p[(mg - 4 * M_CH) * KDIM + ci * 9 + tap];
    Apack[o] = f2bf(v);
}

// 4x4 average pool: x[8,256,64,64] -> xd[8,256,16,16]
__global__ void pool4_k(const float* __restrict__ x, float* __restrict__ xd) {
    int i = blockIdx.x * 256 + threadIdx.x;          // [b][c][ph][pw]
    if (i >= BN_ * CIN * 16 * 16) return;
    int pw = i & 15, ph = (i >> 4) & 15, c = (i >> 8) & 255, b = i >> 16;
    const float* p = x + ((b * CIN + c) * HW + ph * 4) * HW + pw * 4;
    float s = 0.f;
    for (int r = 0; r < 4; ++r)
        for (int q = 0; q < 4; ++q) s += p[r * HW + q];
    xd[i] = s * 0.0625f;
}

// Router conv3x3 256->32 on 16x16 + BN + SiLU (tiny, direct)
__global__ void router_conv_k(const float* __restrict__ xd,
                              const float* __restrict__ Wr1,
                              const float* __restrict__ sr1,
                              const float* __restrict__ br1,
                              float* __restrict__ hbuf) {
    int i = blockIdx.x * 256 + threadIdx.x;          // [b][oc][ph][pw], 65536 total
    if (i >= BN_ * 32 * 16 * 16) return;
    int pw = i & 15, ph = (i >> 4) & 15, oc = (i >> 8) & 31, b = i >> 13;
    const float* w = Wr1 + oc * CIN * 9;
    float s = 0.f;
    for (int ci = 0; ci < CIN; ++ci) {
        const float* xp = xd + (b * CIN + ci) * 256;
        for (int kh = 0; kh < 3; ++kh) {
            int hh = ph + kh - 1; if (hh < 0 || hh > 15) continue;
            for (int kw = 0; kw < 3; ++kw) {
                int ww = pw + kw - 1; if (ww < 0 || ww > 15) continue;
                s += xp[hh * 16 + ww] * w[ci * 9 + kh * 3 + kw];
            }
        }
    }
    hbuf[i] = silu_f(s * sr1[oc] + br1[oc]);
}

// Global mean -> 1x1 conv -> BN -> softmax -> top-2 renorm -> w_full[b][5]
__global__ void router_weights_k(const float* __restrict__ hbuf,
                                 const float* __restrict__ Wr2,
                                 const float* __restrict__ sr2,
                                 const float* __restrict__ br2,
                                 float* __restrict__ wfull) {
    __shared__ float hbar[BN_][32];
    int t = threadIdx.x;                              // 256 = 8*32
    int b = t >> 5, c = t & 31;
    const float* p = hbuf + (b * 32 + c) * 256;
    float s = 0.f;
    for (int i = 0; i < 256; ++i) s += p[i];
    hbar[b][c] = s * (1.0f / 256.0f);
    __syncthreads();
    if (t < BN_) {
        b = t;
        float lg[4];
        for (int e = 0; e < 4; ++e) {
            float d = 0.f;
            for (int c2 = 0; c2 < 32; ++c2) d += Wr2[e * 32 + c2] * hbar[b][c2];
            lg[e] = d * sr2[e] + br2[e];
        }
        float mx = fmaxf(fmaxf(lg[0], lg[1]), fmaxf(lg[2], lg[3]));
        float w[4], sum = 0.f;
        for (int e = 0; e < 4; ++e) { w[e] = __expf(lg[e] - mx); sum += w[e]; }
        for (int e = 0; e < 4; ++e) w[e] /= sum;
        int i1 = 0;
        for (int e = 1; e < 4; ++e) if (w[e] > w[i1]) i1 = e;
        int i2 = -1;
        for (int e = 0; e < 4; ++e) { if (e == i1) continue; if (i2 < 0 || w[e] > w[i2]) i2 = e; }
        float norm = w[i1] + w[i2] + 1e-6f;
        for (int e = 0; e < 4; ++e) wfull[b * NEXP + e] = 0.f;
        wfull[b * NEXP + i1] = w[i1] / norm;
        wfull[b * NEXP + i2] = w[i2] / norm;
        wfull[b * NEXP + 4]  = 1.0f;                  // shared expert weight
    }
}

// ================= fused ghost-expert kernel (WMMA) =================
// Grid: (band=16, batch=8, expert=5). Block: 256 threads = 8 wave32,
// tiled as 4 M-slots x 2 N-slots; each wave owns 2 M-tiles x 4 N-tiles
// -> per K-step: 8 WMMA from 4 A-loads + 8 B-loads (1.5 ds/wmma).
// y1 = SiLU(BN(conv3x3)) for 6 rows x 64 cols via implicit GEMM
// M=128, N=384, K=2304 (tap-major K; channels-last bf16 xT B operand).
__launch_bounds__(256, 1)
__global__ void ghost_expert_wmma_k(
    const unsigned short* __restrict__ xT,           // [b][h][w][ci] bf16
    const unsigned short* __restrict__ Apack,
    const float* __restrict__ se_p, const float* __restrict__ be_p,   // (4,128)
    const float* __restrict__ ss_p, const float* __restrict__ bs_p,   // (128)
    const float* __restrict__ We_c, const float* __restrict__ se_c, const float* __restrict__ be_c,
    const float* __restrict__ Ws_c, const float* __restrict__ ss_c, const float* __restrict__ bs_c,
    const float* __restrict__ wfull,
    float* __restrict__ out)
{
    // ~135 KB static LDS (CDNA5: up to 320 KB per workgroup)
    __shared__ __attribute__((aligned(16))) unsigned short lds_y1[M_CH * NPOS];   // 96 KB bf16 [m][pos]
    __shared__ __attribute__((aligned(16))) unsigned short lds_A[M_CH * KCHUNK];  // 16 KB [m][kk]
    __shared__ __attribute__((aligned(16))) unsigned short lds_B[128 * KCHUNK];   // 16 KB [pl][kk]
    __shared__ float lds_wc[M_CH * 9];
    __shared__ float lds_sc[M_CH], lds_bc[M_CH], lds_sp[M_CH], lds_bp[M_CH];

    const int band = blockIdx.x;           // 0..15
    const int b    = blockIdx.y;           // 0..7
    const int e    = blockIdx.z;           // 0..4 (4 == shared expert)
    const int h0   = band * OUTROWS;
    const int t    = threadIdx.x;
    const int wave = t >> 5, lane = t & 31;
    const int g = lane >> 4, mr = lane & 15;
    const int mslot = wave & 3;            // 4 M-slots x 2 M-tiles = 128 channels
    const int nslot = wave >> 2;           // 2 N-slots x 4 N-tiles = 128 positions

    const float wexp = wfull[b * NEXP + e];
    if (wexp == 0.0f) return;              // skip non-top-k experts (uniform branch)

    const float* sp = (e < 4) ? (se_p + e * M_CH)     : ss_p;
    const float* bp = (e < 4) ? (be_p + e * M_CH)     : bs_p;
    const float* wc = (e < 4) ? (We_c + e * M_CH * 9) : Ws_c;
    const float* sc = (e < 4) ? (se_c + e * M_CH)     : ss_c;
    const float* bc = (e < 4) ? (be_c + e * M_CH)     : bs_c;

    for (int i = t; i < M_CH * 9; i += 256) lds_wc[i] = wc[i];
    for (int i = t; i < M_CH; i += 256) {
        lds_sc[i] = sc[i]; lds_bc[i] = bc[i];
        lds_sp[i] = sp[i]; lds_bp[i] = bp[i];
    }

    const unsigned short* Aw  = Apack + e * M_CH * KDIM;
    const unsigned short* xTb = xT + (size_t)b * (HW * HW) * CIN;

    // staging thread mappings (loop-invariant)
    const int am = t >> 1, akk = (t & 1) * 32;       // A: 32 bf16 (64 B) per thread
    const int pl = t >> 1, pkk = (t & 1) * 32;       // B: 32 ci (64 B) per thread
    const int prow = pl >> 6, pcol = pl & 63;        // B position -> (row, col)

    for (int pass = 0; pass < NPASS; ++pass) {
        const int p0 = pass * 128;         // two y1 rows (8 n-tiles) per pass
        v8f acc[2][4] = {};

        for (int k0 = 0; k0 < KDIM; k0 += KCHUNK) {
            // ---- stage A chunk: 128 rows x 64 k (bf16), 64 B / thread ----
            {
                const uint4* srcp = (const uint4*)(Aw + am * KDIM + k0 + akk);
                uint4 d0 = srcp[0], d1 = srcp[1], d2 = srcp[2], d3 = srcp[3];
                uint4* dst = (uint4*)&lds_A[am * KCHUNK + akk];
                dst[0] = d0; dst[1] = d1; dst[2] = d2; dst[3] = d3;
                __builtin_prefetch((const void*)(Aw + am * KDIM + k0 + KCHUNK + akk), 0, 1);
            }
            // ---- stage B chunk: 128 positions x 64 ci; 64 B / thread ----
            {
                const int tap = k0 >> 8;           // chunk-uniform (64 | 256)
                const int kh  = tap / 3;
                const int kw  = tap - kh * 3;
                const int ci0 = k0 & 255;
                const int hh  = h0 - 2 + pass * 2 + prow + kh;  // x row
                const int col = pcol + kw - 1;                  // x column
                uint4 d0 = {0,0,0,0}, d1 = {0,0,0,0}, d2 = {0,0,0,0}, d3 = {0,0,0,0};
                if (hh >= 0 && hh < HW && col >= 0 && col < HW) {
                    const uint4* src =
                        (const uint4*)(xTb + ((size_t)(hh * HW + col)) * CIN + ci0 + pkk);
                    d0 = src[0]; d1 = src[1]; d2 = src[2]; d3 = src[3];
                }
                uint4* dst = (uint4*)&lds_B[pl * KCHUNK + pkk];
                dst[0] = d0; dst[1] = d1; dst[2] = d2; dst[3] = d3;
            }
            __syncthreads();

            // ---- 2 K-steps x (2 M-tiles x 4 N-tiles) = 16 WMMA / barrier pair ----
            #pragma unroll
            for (int s = 0; s < 2; ++s) {
                const int kbase = s * 32;
                v16bf av[2];
                #pragma unroll
                for (int mi = 0; mi < 2; ++mi) {
                    const int mrow = (mslot * 2 + mi) * 16 + mr;
                    v8bf alo = *(const v8bf*)&lds_A[mrow * KCHUNK + kbase + g * 8];
                    v8bf ahi = *(const v8bf*)&lds_A[mrow * KCHUNK + kbase + 16 + g * 8];
                    av[mi] = __builtin_shufflevector(alo, ahi, 0,1,2,3,4,5,6,7,8,9,10,11,12,13,14,15);
                }
                v16bf bv[4];
                #pragma unroll
                for (int nt = 0; nt < 4; ++nt) {
                    const int nrow = (nslot * 4 + nt) * 16 + mr;
                    v8bf blo = *(const v8bf*)&lds_B[nrow * KCHUNK + kbase + g * 8];
                    v8bf bhi = *(const v8bf*)&lds_B[nrow * KCHUNK + kbase + 16 + g * 8];
                    bv[nt] = __builtin_shufflevector(blo, bhi, 0,1,2,3,4,5,6,7,8,9,10,11,12,13,14,15);
                }
                #pragma unroll
                for (int mi = 0; mi < 2; ++mi)
                    #pragma unroll
                    for (int nt = 0; nt < 4; ++nt)
                        acc[mi][nt] = __builtin_amdgcn_wmma_f32_16x16x32_bf16(
                            false, av[mi], false, bv[nt], (short)0, acc[mi][nt], false, false);
            }
            __syncthreads();
        }

        // ---- epilogue: BN + SiLU, spill y1 tile to LDS as bf16 ----
        #pragma unroll
        for (int mi = 0; mi < 2; ++mi) {
            #pragma unroll
            for (int nt = 0; nt < 4; ++nt) {
                int pos = p0 + (nslot * 4 + nt) * 16 + mr;
                #pragma unroll
                for (int r = 0; r < 8; ++r) {
                    int m = (mslot * 2 + mi) * 16 + g * 8 + r;  // C/D: M = r + 8*(lane/16)
                    float v = acc[mi][nt][r] * lds_sp[m] + lds_bp[m];
                    lds_y1[m * NPOS + pos] = f2bf(silu_f(v));
                }
            }
        }
    }
    __syncthreads();

    // ---- depthwise 3x3 + BN + SiLU + weighted combine (1 thread = 1 out pos) ----
    const int qr = t >> 6, col = t & 63;           // 4 rows x 64 cols
    const int h = h0 + qr;
    float* outb = out + (size_t)b * COUT * HW * HW;
    const int oidx0 = h * HW + col;

    for (int m = 0; m < M_CH; ++m) {
        // first half of output channels = y1 itself
        float y = bf2f(lds_y1[m * NPOS + (qr + 1) * 64 + col]);
        atomicAdd(&outb[m * (HW * HW) + oidx0], wexp * y);

        // second half = SiLU(BN(depthwise3x3(y1))); zero-pad outside image
        float s = 0.0f;
        #pragma unroll
        for (int dr = 0; dr < 3; ++dr) {
            int hh = h + dr - 1; if (hh < 0 || hh >= HW) continue;
            int yrow = hh - h0 + 1;                // always 0..5 inside tile
            #pragma unroll
            for (int dc = 0; dc < 3; ++dc) {
                int ww = col + dc - 1; if (ww < 0 || ww >= HW) continue;
                s += bf2f(lds_y1[m * NPOS + yrow * 64 + ww]) * lds_wc[m * 9 + dr * 3 + dc];
            }
        }
        s = silu_f(s * lds_sc[m] + lds_bc[m]);
        atomicAdd(&outb[(M_CH + m) * (HW * HW) + oidx0], wexp * s);
    }
}

// ================= host launch =================
extern "C" void kernel_launch(void* const* d_in, const int* in_sizes, int n_in,
                              void* d_out, int out_size, void* d_ws, size_t ws_size,
                              hipStream_t stream) {
    const float* x    = (const float*)d_in[0];
    const float* Wr1  = (const float*)d_in[1];
    const float* sr1  = (const float*)d_in[2];
    const float* br1  = (const float*)d_in[3];
    const float* Wr2  = (const float*)d_in[4];
    const float* sr2  = (const float*)d_in[5];
    const float* br2  = (const float*)d_in[6];
    const float* We_p = (const float*)d_in[7];
    const float* se_p = (const float*)d_in[8];
    const float* be_p = (const float*)d_in[9];
    const float* We_c = (const float*)d_in[10];
    const float* se_c = (const float*)d_in[11];
    const float* be_c = (const float*)d_in[12];
    const float* Ws_p = (const float*)d_in[13];
    const float* ss_p = (const float*)d_in[14];
    const float* bs_p = (const float*)d_in[15];
    const float* Ws_c = (const float*)d_in[16];
    const float* ss_c = (const float*)d_in[17];
    const float* bs_c = (const float*)d_in[18];
    float* out = (float*)d_out;

    // workspace layout (~22 MB)
    char* ws = (char*)d_ws;
    unsigned short* Apack = (unsigned short*)ws;                        // 5*128*2304 bf16
    unsigned short* xT = Apack + (size_t)NEXP * M_CH * KDIM;            // 8*64*64*256 bf16
    float* xd    = (float*)(xT + (size_t)BN_ * HW * HW * CIN);          // 8*256*16*16 f32
    float* hbuf  = xd + BN_ * CIN * 16 * 16;                            // 8*32*16*16
    float* wfull = hbuf + BN_ * 32 * 16 * 16;                           // 8*5

    zero_out_k<<<(out_size + 255) / 256, 256, 0, stream>>>(out, out_size);
    xpose_k<<<(BN_ * HW * HW * CIN) / 256, 256, 0, stream>>>(x, xT);
    pack_weights_k<<<(NEXP * M_CH * KDIM + 255) / 256, 256, 0, stream>>>(We_p, Ws_p, Apack);
    pool4_k<<<(BN_ * CIN * 16 * 16) / 256, 256, 0, stream>>>(x, xd);
    router_conv_k<<<(BN_ * 32 * 16 * 16) / 256, 256, 0, stream>>>(xd, Wr1, sr1, br1, hbuf);
    router_weights_k<<<1, 256, 0, stream>>>(hbuf, Wr2, sr2, br2, wfull);

    dim3 grid(16, BN_, NEXP);
    ghost_expert_wmma_k<<<grid, 256, 0, stream>>>(xT, Apack, se_p, be_p, ss_p, bs_p,
        We_c, se_c, be_c, Ws_c, ss_c, bs_c, wfull, out);
}
